// SelfAttention_12945031430761
// MI455X (gfx1250) — matrix operand order repaired and checked by reading, not compiled, not run on hardware
//
#include <hip/hip_runtime.h>
#include <hip/hip_bf16.h>

#define DIM     768
#define NHEADS  12
#define HD      64
#define SEQ     2048
#define BATCH   4
#define ROWS    (BATCH*SEQ)        // 8192
#define ATTN_SCALE 0.125f          // 64^-0.5

typedef __attribute__((ext_vector_type(16))) __bf16 v16bf;
typedef __attribute__((ext_vector_type(8)))  __bf16 v8bf;
typedef __attribute__((ext_vector_type(8)))  float  v8f;

union FragA { v16bf v; v8bf h[2]; };
union PackB { v8bf v; __bf16 b[8]; };

__device__ __forceinline__ v8f wmma_bf16(v16bf a, v16bf b, v8f c) {
  // (neg_a, A, neg_b, B, c_mod, C, reuse_a, reuse_b)
  return __builtin_amdgcn_wmma_f32_16x16x32_bf16(false, a, false, b, (short)0, c, false, false);
}

// ---------------------------------------------------------------- convert
__global__ void cvt_bf16(const float* __restrict__ in, __bf16* __restrict__ out, int n) {
  int i = blockIdx.x * 256 + threadIdx.x;
  if (i < n) out[i] = (__bf16)in[i];
}

// w[768][N] f32  ->  wt[N][768] bf16   (one-time, 32x32 LDS tiles)
__global__ void __launch_bounds__(256) transpose_w_bf16(
    const float* __restrict__ w, __bf16* __restrict__ wt, int N) {
  __shared__ __bf16 Tl[32][33];          // +1 pad: conflict-free transpose
  const int kt = blockIdx.y * 32;
  const int nt = blockIdx.x * 32;
  const int t  = threadIdx.x;
  const int r  = t >> 3;                 // 0..31
  const int c0 = (t & 7) * 4;            // 0,4,...,28
  const float* src = w + (size_t)(kt + r) * N + nt + c0;
  #pragma unroll
  for (int i = 0; i < 4; i++) Tl[c0 + i][r] = (__bf16)src[i];
  __syncthreads();
  __bf16* dst = wt + (size_t)(nt + r) * DIM + kt + c0;
  #pragma unroll
  for (int i = 0; i < 4; i++) dst[i] = Tl[r][c0 + i];
}

// ---------------------------------------------------------------- GEMM
// C[ROWS,N] = A[ROWS,768](bf16) @ W(bf16, pre-transposed as Wt[N][768]) + bias
// mode 0: N=2304 -> Q(*scale)/K bf16 [B,H,N,hd], V bf16 transposed [B,H,hd,N]
// mode 1: N=768  -> fp32 row-major output
__global__ void __launch_bounds__(256) gemm_qkv_proj(
    const __bf16* __restrict__ A, const __bf16* __restrict__ Wt,
    const float* __restrict__ bias, int N, int mode,
    __bf16* __restrict__ qo, __bf16* __restrict__ ko, __bf16* __restrict__ vo,
    float* __restrict__ fo)
{
  __shared__ __align__(16) __bf16 Al[128][32];   // [row][k]
  __shared__ __align__(16) __bf16 Wl[128][32];   // [col][k]

  const int tid  = threadIdx.x;
  const int lane = tid & 31;
  const int w    = tid >> 5;
  const int wr   = w >> 1;           // 0..3 : 32-row band
  const int wc   = w & 1;            // 0..1 : 64-col band
  const int cn   = lane & 15;
  const int hi   = lane >> 4;
  const int r0   = hi ? 8 : 0;

  const int gM0 = blockIdx.y * 128;
  const int gN0 = blockIdx.x * 128;

  v8f acc[2][4];
  #pragma unroll
  for (int i = 0; i < 2; i++)
    #pragma unroll
    for (int j = 0; j < 4; j++) acc[i][j] = {};

  const int row = tid >> 1, half = tid & 1;      // staging coords (b128 copies)
  for (int kb = 0; kb < DIM; kb += 32) {
    __syncthreads();
    {
      const uint4* sa = (const uint4*)(A  + (size_t)(gM0 + row) * DIM + kb + half * 16);
      uint4* da = (uint4*)(&Al[row][half * 16]);
      da[0] = sa[0]; da[1] = sa[1];
      const uint4* sw = (const uint4*)(Wt + (size_t)(gN0 + row) * DIM + kb + half * 16);
      uint4* dw = (uint4*)(&Wl[row][half * 16]);
      dw[0] = sw[0]; dw[1] = sw[1];
    }
    __syncthreads();

    FragA a[2], b[4];
    #pragma unroll
    for (int ms = 0; ms < 2; ms++) {
      int rr = 32 * wr + 16 * ms + cn;
      int k0 = hi ? 8 : 0;                       // A layout: {0-7,16-23}/{8-15,24-31}
      a[ms].h[0] = *(const v8bf*)&Al[rr][k0];
      a[ms].h[1] = *(const v8bf*)&Al[rr][k0 + 16];
    }
    #pragma unroll
    for (int ns = 0; ns < 4; ns++) {
      int col = 64 * wc + 16 * ns + cn;
      int k0  = hi ? 8 : 0;                      // B layout: {0-7,16-23}/{8-15,24-31}
      b[ns].h[0] = *(const v8bf*)&Wl[col][k0];
      b[ns].h[1] = *(const v8bf*)&Wl[col][k0 + 16];
    }
    #pragma unroll
    for (int ms = 0; ms < 2; ms++)
      #pragma unroll
      for (int ns = 0; ns < 4; ns++)
        acc[ms][ns] = wmma_bf16(a[ms].v, b[ns].v, acc[ms][ns]);
  }

  #pragma unroll
  for (int ms = 0; ms < 2; ms++) {
    #pragma unroll
    for (int ns = 0; ns < 4; ns++) {
      int col = gN0 + 64 * wc + 16 * ns + cn;
      float bv = bias[col];
      if (mode == 0) {
        int which = col / DIM;
        int rem   = col - which * DIM;
        int head  = rem >> 6;
        int dd    = rem & 63;
        int row0  = gM0 + 32 * wr + 16 * ms + r0;
        int bb = row0 >> 11, nn0 = row0 & 2047;
        if (which == 2) {                         // V: transposed, vector store
          PackB pk;
          #pragma unroll
          for (int j = 0; j < 8; j++) pk.b[j] = (__bf16)(acc[ms][ns][j] + bv);
          *(v8bf*)&vo[((size_t)(bb * NHEADS + head) * HD + dd) * SEQ + nn0] = pk.v;
        } else {
          __bf16* dst = (which == 0) ? qo : ko;
          float scl   = (which == 0) ? ATTN_SCALE : 1.0f;
          #pragma unroll
          for (int j = 0; j < 8; j++) {
            size_t idx = ((size_t)(bb * NHEADS + head) * SEQ + nn0 + j) * HD + dd;
            dst[idx] = (__bf16)((acc[ms][ns][j] + bv) * scl);
          }
        }
      } else {
        #pragma unroll
        for (int j = 0; j < 8; j++) {
          int rr = gM0 + 32 * wr + 16 * ms + r0 + j;
          fo[(size_t)rr * DIM + col] = acc[ms][ns][j] + bv;
        }
      }
    }
  }
}

// ---------------------------------------------------------------- attention
// Q,K bf16 [B,H,SEQ,HD] (Q pre-scaled); V bf16 transposed [B,H,HD,SEQ].
// O bf16 [B,SEQ,DIM] with col = h*64+d.  64-key chunks, 4 waves x 16 queries.
__global__ void __launch_bounds__(128) flash_attn(
    const __bf16* __restrict__ Q, const __bf16* __restrict__ K,
    const __bf16* __restrict__ V, __bf16* __restrict__ O)
{
  __shared__ __align__(16) __bf16 Kl[64][64];     // [key][d]
  __shared__ __align__(16) __bf16 Vl[64][64];     // [d][key]
  __shared__ __align__(16) float  Sl[4][16][64];  // per-wave scores
  __shared__ __align__(16) __bf16 Pl[4][16][64];  // per-wave probs
  __shared__ float Mx[4][2][16];                  // partial row max
  __shared__ float Sm[4][2][16];                  // partial row sum
  __shared__ float Cl[4][16];                     // per-row correction
  __shared__ float Ll[4][16];                     // per-row 1/l

  const int tid  = threadIdx.x;
  const int lane = tid & 31;
  const int w    = tid >> 5;
  const int cn   = lane & 15;
  const int hi   = lane >> 4;
  const int r0   = hi ? 8 : 0;

  const int blk = blockIdx.x;
  const int qt  = blk & 31;                 // 32 query blocks of 64
  const int hh  = (blk >> 5) % NHEADS;
  const int bb  = blk / (32 * NHEADS);

  const size_t head_base = (size_t)(bb * NHEADS + hh) * SEQ * HD;
  const int q0 = qt * 64 + w * 16;          // wave's first query

  FragA qa[2];                              // Q 16x64 as two A-fragments
  #pragma unroll
  for (int ds = 0; ds < 2; ds++) {
    int doff = ds * 32 + (hi ? 8 : 0);
    const __bf16* qp = Q + head_base + (size_t)(q0 + cn) * HD + doff;
    qa[ds].h[0] = *(const v8bf*)(qp);
    qa[ds].h[1] = *(const v8bf*)(qp + 16);
  }

  v8f o[4];
  #pragma unroll
  for (int t = 0; t < 4; t++) o[t] = {};
  float m = -INFINITY, l = 0.0f;

  const int srow_ = tid >> 1, shalf = tid & 1;    // staging coords

  for (int kc = 0; kc < SEQ; kc += 64) {
    __syncthreads();                        // previous chunk fully consumed
    {   // K: 64 keys x 64 d row-major; V: 64 d x 64 keys (pre-transposed)
      const uint4* ks = (const uint4*)(K + head_base + (size_t)(kc + srow_) * HD + shalf * 32);
      uint4* kd = (uint4*)&Kl[srow_][shalf * 32];
      kd[0] = ks[0]; kd[1] = ks[1]; kd[2] = ks[2]; kd[3] = ks[3];
      const uint4* vs = (const uint4*)(V + head_base + (size_t)srow_ * SEQ + kc + shalf * 32);
      uint4* vd = (uint4*)&Vl[srow_][shalf * 32];
      vd[0] = vs[0]; vd[1] = vs[1]; vd[2] = vs[2]; vd[3] = vs[3];
    }
    __syncthreads();

    // S = Q K^T : 4 key subtiles, K-dim 64 = 2 WMMA steps each
    v8f s_acc[4];
    #pragma unroll
    for (int s = 0; s < 4; s++) s_acc[s] = {};
    #pragma unroll
    for (int ds = 0; ds < 2; ds++) {
      int doff = ds * 32 + (hi ? 8 : 0);
      #pragma unroll
      for (int s = 0; s < 4; s++) {
        FragA kb;
        kb.h[0] = *(const v8bf*)&Kl[16 * s + cn][doff];
        kb.h[1] = *(const v8bf*)&Kl[16 * s + cn][doff + 16];
        s_acc[s] = wmma_bf16(qa[ds].v, kb.v, s_acc[s]);
      }
    }

    #pragma unroll
    for (int s = 0; s < 4; s++)             // spill C-layout tiles to LDS
      #pragma unroll
      for (int j = 0; j < 8; j++)
        Sl[w][r0 + j][16 * s + cn] = s_acc[s][j];
    asm volatile("s_wait_dscnt 0" ::: "memory");

    // online softmax: row = cn, this lane handles keys [hi*32, hi*32+32)
    float sv[32];
    {
      const float4* sp = (const float4*)&Sl[w][cn][hi * 32];
      #pragma unroll
      for (int g = 0; g < 8; g++) {
        float4 f = sp[g];
        sv[4*g] = f.x; sv[4*g+1] = f.y; sv[4*g+2] = f.z; sv[4*g+3] = f.w;
      }
    }
    float pmax = sv[0];
    #pragma unroll
    for (int j = 1; j < 32; j++) pmax = fmaxf(pmax, sv[j]);
    Mx[w][hi][cn] = pmax;
    asm volatile("s_wait_dscnt 0" ::: "memory");
    float mnew = fmaxf(m, fmaxf(pmax, Mx[w][1 - hi][cn]));
    float corr = __expf(m - mnew);

    float sum = 0.0f;
    #pragma unroll
    for (int g = 0; g < 4; g++) {
      PackB pk;
      #pragma unroll
      for (int j = 0; j < 8; j++) {
        float p = __expf(sv[8 * g + j] - mnew);
        sum += p;
        pk.b[j] = (__bf16)p;
      }
      *(v8bf*)&Pl[w][cn][hi * 32 + 8 * g] = pk.v;
    }
    Sm[w][hi][cn] = sum;
    if (!hi) Cl[w][cn] = corr;
    asm volatile("s_wait_dscnt 0" ::: "memory");
    l = l * corr + sum + Sm[w][1 - hi][cn];
    m = mnew;

    {   // rescale O accumulators (row -> VGPR index in C layout)
      float c8[8];
      #pragma unroll
      for (int j = 0; j < 8; j++) c8[j] = Cl[w][r0 + j];
      #pragma unroll
      for (int t = 0; t < 4; t++)
        #pragma unroll
        for (int j = 0; j < 8; j++) o[t][j] *= c8[j];
    }

    // O += P @ V : P 16x64 = two A-frags; V 64x(4x16) from Vl[d][key]
    FragA pf[2];
    #pragma unroll
    for (int kk = 0; kk < 2; kk++) {
      int k0 = kk * 32 + (hi ? 8 : 0);
      pf[kk].h[0] = *(const v8bf*)&Pl[w][cn][k0];
      pf[kk].h[1] = *(const v8bf*)&Pl[w][cn][k0 + 16];
    }
    #pragma unroll
    for (int t = 0; t < 4; t++) {
      #pragma unroll
      for (int kk = 0; kk < 2; kk++) {
        FragA vb;
        int k0 = kk * 32 + (hi ? 8 : 0);
        vb.h[0] = *(const v8bf*)&Vl[16 * t + cn][k0];
        vb.h[1] = *(const v8bf*)&Vl[16 * t + cn][k0 + 16];
        o[t] = wmma_bf16(pf[kk].v, vb.v, o[t]);
      }
    }
  }

  if (!hi) Ll[w][cn] = 1.0f / l;
  asm volatile("s_wait_dscnt 0" ::: "memory");
  float li[8];
  #pragma unroll
  for (int j = 0; j < 8; j++) li[j] = Ll[w][r0 + j];

  #pragma unroll
  for (int t = 0; t < 4; t++) {
    int col = hh * HD + 16 * t + cn;
    #pragma unroll
    for (int j = 0; j < 8; j++) {
      int nn = q0 + r0 + j;
      O[((size_t)bb * SEQ + nn) * DIM + col] = (__bf16)(o[t][j] * li[j]);
    }
  }
}

// ---------------------------------------------------------------- launch
extern "C" void kernel_launch(void* const* d_in, const int* in_sizes, int n_in,
                              void* d_out, int out_size, void* d_ws, size_t ws_size,
                              hipStream_t stream) {
  const float* x      = (const float*)d_in[0];
  const float* w_qkv  = (const float*)d_in[1];
  const float* b_qkv  = (const float*)d_in[2];
  const float* w_proj = (const float*)d_in[3];
  const float* b_proj = (const float*)d_in[4];
  float* out = (float*)d_out;

  char* ws = (char*)d_ws;                       // all offsets 256B-aligned
  __bf16* xb   = (__bf16*)(ws);                 // 8192x768        : 12,582,912 B
  __bf16* wqt  = (__bf16*)(ws + 12582912);      // [2304][768]     :  3,538,944 B
  __bf16* wpt  = (__bf16*)(ws + 16121856);      // [768][768]      :  1,179,648 B
  __bf16* Qb   = (__bf16*)(ws + 17301504);      // [B,H,N,hd]      : 12,582,912 B
  __bf16* Kb   = (__bf16*)(ws + 29884416);      // [B,H,N,hd]
  __bf16* Vtb  = (__bf16*)(ws + 42467328);      // [B,H,hd,N] (transposed)
  __bf16* Ab   = (__bf16*)(ws + 55050240);      // attn out [8192,768] bf16

  cvt_bf16<<<(ROWS * DIM + 255) / 256, 256, 0, stream>>>(x, xb, ROWS * DIM);
  transpose_w_bf16<<<dim3(3 * DIM / 32, DIM / 32), 256, 0, stream>>>(w_qkv, wqt, 3 * DIM);
  transpose_w_bf16<<<dim3(DIM / 32, DIM / 32), 256, 0, stream>>>(w_proj, wpt, DIM);

  dim3 g1(3 * DIM / 128, ROWS / 128);
  gemm_qkv_proj<<<g1, 256, 0, stream>>>(xb, wqt, b_qkv, 3 * DIM, 0, Qb, Kb, Vtb, nullptr);

  flash_attn<<<BATCH * NHEADS * (SEQ / 64), 128, 0, stream>>>(Qb, Kb, Vtb, Ab);

  dim3 g2(DIM / 128, ROWS / 128);
  gemm_qkv_proj<<<g2, 256, 0, stream>>>(Ab, wpt, b_proj, DIM, 1, nullptr, nullptr, nullptr, out);
}
